// MambaTRM_82609400971398
// MI455X (gfx1250) — compile-verified, matrix operand order checked by reading, not run on hardware
//
#include <hip/hip_runtime.h>
#include <hip/hip_bf16.h>
#include <math.h>

#define DM   384      // d_model == d_inner
#define DM2  768      // 2*DM
#define DS   4        // d_state
#define DH   128      // probe head dim

typedef __attribute__((ext_vector_type(16))) _Float16 v16h;
typedef __attribute__((ext_vector_type(8)))  _Float16 v8h;
typedef __attribute__((ext_vector_type(8)))  float    v8f;

__device__ __forceinline__ float siluf(float v)     { return v / (1.0f + __expf(-v)); }
__device__ __forceinline__ float softplusf(float v) { return (v > 20.0f) ? v : log1pf(__expf(v)); }
__device__ __forceinline__ float geluf(float v)     { return 0.5f * v * (1.0f + erff(v * 0.70710678118654752f)); }

// ---------------- f32 -> f16 conversion ----------------
__global__ void f32_to_f16_kernel(const float* __restrict__ s, _Float16* __restrict__ d, size_t n) {
    size_t i = (size_t)blockIdx.x * 256 + threadIdx.x;
    if (i < n) d[i] = (_Float16)s[i];
}

// ---------------- build concat(events, type_emb) in f16 ----------------
__global__ void embed_kernel(const float* __restrict__ ev, const int* __restrict__ ty,
                             const float* __restrict__ temb, _Float16* __restrict__ out, int rows) {
    size_t i = (size_t)blockIdx.x * 256 + threadIdx.x;
    if (i >= (size_t)rows * DM2) return;
    int row = (int)(i / DM2), j = (int)(i % DM2);
    float v = (j < DM) ? ev[(size_t)row * DM + j] : temb[ty[row] * DM + (j - DM)];
    out[i] = (_Float16)v;
}

// ---------------- WMMA fragment loader: 16 halves per lane (two 16B chunks) ----------------
__device__ __forceinline__ v16h load_frag(const _Float16* p) {
    v8h x0 = *(const v8h*)(p);
    v8h x1 = *(const v8h*)(p + 16);
    v16h r;
#pragma unroll
    for (int i = 0; i < 8; ++i) { r[i] = x0[i]; r[i + 8] = x1[i]; }
    return r;
}

// ---------------- WMMA GEMM: C[M,N] = A[M,K] @ W[N,K]^T (+bias, +resid, act) ----------------
// One wave computes a 32x32 output block (2x2 WMMA tiles, A/B fragment reuse);
// 8 waves per 256-thread block stacked along M -> 256 rows per block.
__global__ void __launch_bounds__(256)
wmma_gemm_kernel(const _Float16* __restrict__ A, const _Float16* __restrict__ W,
                 float* __restrict__ C, const float* __restrict__ bias,
                 const float* __restrict__ resid, int M, int N, int K, int act) {
    const int lane = threadIdx.x & 31;
    const int wv   = threadIdx.x >> 5;
    const int tn   = blockIdx.x * 32;
    const int tm   = (blockIdx.y * 8 + wv) * 32;
    if (tm >= M) return;                    // wave-uniform
    const int r15 = lane & 15;
    const int kb  = (lane >> 4) << 3;       // 0 or 8 (K-subblock select per ISA layout)
    const _Float16* ap0 = A + (size_t)(tm + r15)      * K + kb;
    const _Float16* ap1 = A + (size_t)(tm + 16 + r15) * K + kb;
    const _Float16* wp0 = W + (size_t)(tn + r15)      * K + kb;
    const _Float16* wp1 = W + (size_t)(tn + 16 + r15) * K + kb;
    v8f acc00 = {}, acc01 = {}, acc10 = {}, acc11 = {};
    for (int kk = 0; kk < K; kk += 32) {
        v16h a0 = load_frag(ap0 + kk);
        v16h a1 = load_frag(ap1 + kk);
        v16h b0 = load_frag(wp0 + kk);
        v16h b1 = load_frag(wp1 + kk);
        acc00 = __builtin_amdgcn_wmma_f32_16x16x32_f16(false, a0, false, b0, (short)0, acc00, false, false);
        acc01 = __builtin_amdgcn_wmma_f32_16x16x32_f16(false, a0, false, b1, (short)0, acc01, false, false);
        acc10 = __builtin_amdgcn_wmma_f32_16x16x32_f16(false, a1, false, b0, (short)0, acc10, false, false);
        acc11 = __builtin_amdgcn_wmma_f32_16x16x32_f16(false, a1, false, b1, (short)0, acc11, false, false);
    }
    // C layout per tile: VGPR r -> row r (lanes 0-15) / row 8+r (lanes 16-31), col = lane&15
    const int rofs = (lane >> 4) << 3;
    const int cn0  = tn + r15;
    const int cn1  = tn + 16 + r15;
    const float bi0 = bias ? bias[cn0] : 0.f;
    const float bi1 = bias ? bias[cn1] : 0.f;
#pragma unroll
    for (int r = 0; r < 8; ++r) {
        const int rm0 = tm + rofs + r;
        const int rm1 = tm + 16 + rofs + r;
        size_t i00 = (size_t)rm0 * N + cn0;
        size_t i01 = (size_t)rm0 * N + cn1;
        size_t i10 = (size_t)rm1 * N + cn0;
        size_t i11 = (size_t)rm1 * N + cn1;
        float v00 = acc00[r] + bi0;
        float v01 = acc01[r] + bi1;
        float v10 = acc10[r] + bi0;
        float v11 = acc11[r] + bi1;
        if (resid) { v00 += resid[i00]; v01 += resid[i01]; v10 += resid[i10]; v11 += resid[i11]; }
        if (act == 1) { v00 = geluf(v00); v01 = geluf(v01); v10 = geluf(v10); v11 = geluf(v11); }
        C[i00] = v00; C[i01] = v01; C[i10] = v10; C[i11] = v11;
    }
}

// ---------------- row LayerNorm over DM, optional f32 / f16 outputs ----------------
__global__ void __launch_bounds__(128)
ln_kernel(const float* __restrict__ x, const float* __restrict__ g, const float* __restrict__ b,
          float* __restrict__ o32, _Float16* __restrict__ o16) {
    __shared__ float red[128];
    const int row = blockIdx.x, tid = threadIdx.x;
    const float* xr = x + (size_t)row * DM;
    float s = 0.f;
    for (int i = tid; i < DM; i += 128) s += xr[i];
    red[tid] = s; __syncthreads();
    for (int st = 64; st > 0; st >>= 1) { if (tid < st) red[tid] += red[tid + st]; __syncthreads(); }
    const float mu = red[0] / (float)DM; __syncthreads();
    float v = 0.f;
    for (int i = tid; i < DM; i += 128) { float d = xr[i] - mu; v += d * d; }
    red[tid] = v; __syncthreads();
    for (int st = 64; st > 0; st >>= 1) { if (tid < st) red[tid] += red[tid + st]; __syncthreads(); }
    const float inv = rsqrtf(red[0] / (float)DM + 1e-5f);
    for (int i = tid; i < DM; i += 128) {
        float y = g[i] * (xr[i] - mu) * inv + b[i];
        if (o32) o32[(size_t)row * DM + i] = y;
        if (o16) o16[(size_t)row * DM + i] = (_Float16)y;
    }
}

// ---------------- depthwise conv (k=2, causal) + SiLU ----------------
__global__ void conv_silu_kernel(const float* __restrict__ xz, const float* __restrict__ cw,
                                 const float* __restrict__ cb, float* __restrict__ xs,
                                 int rows, int L) {
    size_t i = (size_t)blockIdx.x * 256 + threadIdx.x;
    if (i >= (size_t)rows * DM) return;
    int row = (int)(i / DM), d = (int)(i % DM);
    int l = row % L;
    float xc = xz[(size_t)row * DM2 + d];
    float xm = (l > 0) ? xz[(size_t)(row - 1) * DM2 + d] : 0.f;
    xs[i] = siluf(xm * cw[d * 2 + 0] + xc * cw[d * 2 + 1] + cb[d]);
}

// ---------------- x_proj: xparams[row, 9] = x_ssm[row,:] @ x_w[9,:]^T ----------------
__global__ void xparams_kernel(const float* __restrict__ xs, const float* __restrict__ xw,
                               float* __restrict__ xp, int rows) {
    size_t i = (size_t)blockIdx.x * 256 + threadIdx.x;
    if (i >= (size_t)rows * 9) return;
    int row = (int)(i / 9), j = (int)(i % 9);
    const float* a = xs + (size_t)row * DM;
    const float* w = xw + j * DM;
    float s = 0.f;
    for (int k = 0; k < DM; ++k) s += a[k] * w[k];
    xp[i] = s;
}

// ---------------- selective scan: one thread per (batch, channel), in-place y over x_ssm ----------------
__global__ void scan_kernel(const float* __restrict__ xp, const float* __restrict__ logA,
                            const float* __restrict__ Dv, float* __restrict__ xs,
                            int Bn, int L) {
    int t = blockIdx.x * blockDim.x + threadIdx.x;
    if (t >= Bn * DM) return;
    int b = t / DM, d = t % DM;
    float As[DS], h[DS];
#pragma unroll
    for (int s = 0; s < DS; ++s) { As[s] = -__expf(logA[d * DS + s]); h[s] = 0.f; }
    const float Dd = Dv[d];
    const size_t base = (size_t)b * L;
    for (int l = 0; l < L; ++l) {
        size_t row = base + l;
        float xv = xs[row * DM + d];
        const float* p = xp + row * 9;
        float delta = softplusf(p[8]);
        float y = Dd * xv;
#pragma unroll
        for (int s = 0; s < DS; ++s) {
            float dA = __expf(delta * As[s]);
            h[s] = dA * h[s] + (delta * p[s]) * xv;
            y += h[s] * p[4 + s];
        }
        xs[row * DM + d] = y;
    }
}

// ---------------- gate: a16 = f16( y * silu(z) ) ----------------
__global__ void gate_kernel(const float* __restrict__ y, const float* __restrict__ xz,
                            _Float16* __restrict__ o16, size_t n) {
    size_t i = (size_t)blockIdx.x * 256 + threadIdx.x;
    if (i >= n) return;
    size_t row = i / DM; int d = (int)(i % DM);
    float z = xz[row * DM2 + DM + d];
    o16[i] = (_Float16)(y[i] * siluf(z));
}

// ---------------- masked max+mean pool over prefix ----------------
__global__ void pool_kernel(const float* __restrict__ xn, const int* __restrict__ plen,
                            float* __restrict__ ctx, int Bn, int L) {
    int t = blockIdx.x * blockDim.x + threadIdx.x;
    if (t >= Bn * DM) return;
    int b = t / DM, d = t % DM;
    int P = plen[b];
    float mx = -INFINITY, sm = 0.f;
    for (int l = 0; l < P; ++l) {
        float v = xn[((size_t)b * L + l) * DM + d];
        mx = fmaxf(mx, v); sm += v;
    }
    ctx[t] = mx + sm / (float)P;
}

// ---------------- one attention probe (block per batch) ----------------
__global__ void __launch_bounds__(128)
probe_kernel(float* __restrict__ ctx, const float* __restrict__ cand,
             const float* __restrict__ qw, const float* __restrict__ kw,
             const float* __restrict__ vw, const float* __restrict__ ow,
             const float* __restrict__ lg, const float* __restrict__ lb, int N) {
    __shared__ float red[128];
    __shared__ float cn[DM], kq[DM], ws[DM];
    __shared__ float Qs[DH], ob[DH];
    __shared__ float sc[64];
    const int b = blockIdx.x, tid = threadIdx.x;
    const float* cx = ctx + b * DM;
    // layernorm of context
    float s = 0.f;
    for (int i = tid; i < DM; i += 128) s += cx[i];
    red[tid] = s; __syncthreads();
    for (int st = 64; st > 0; st >>= 1) { if (tid < st) red[tid] += red[tid + st]; __syncthreads(); }
    float mu = red[0] / (float)DM; __syncthreads();
    float v = 0.f;
    for (int i = tid; i < DM; i += 128) { float d = cx[i] - mu; v += d * d; }
    red[tid] = v; __syncthreads();
    for (int st = 64; st > 0; st >>= 1) { if (tid < st) red[tid] += red[tid + st]; __syncthreads(); }
    float inv = rsqrtf(red[0] / (float)DM + 1e-5f);
    for (int i = tid; i < DM; i += 128) cn[i] = lg[i] * (cx[i] - mu) * inv + lb[i];
    __syncthreads();
    // Q[t] = cn . qw[t,:]
    { float q = 0.f; const float* w = qw + tid * DM;
      for (int j = 0; j < DM; ++j) q += cn[j] * w[j];
      Qs[tid] = q; }
    __syncthreads();
    // kq[j] = sum_t Q[t]*kw[t,j]  (so score_n = kq . cand_n)
    for (int j = tid; j < DM; j += 128) {
        float a = 0.f;
        for (int t = 0; t < DH; ++t) a += Qs[t] * kw[t * DM + j];
        kq[j] = a;
    }
    __syncthreads();
    const float* cb = cand + (size_t)b * N * DM;
    if (tid < N) {
        float a = 0.f; const float* c = cb + tid * DM;
        for (int j = 0; j < DM; ++j) a += kq[j] * c[j];
        sc[tid] = a * 0.08838834764831845f;   // 1/sqrt(128)
    }
    __syncthreads();
    if (tid == 0) {
        float mx = -INFINITY;
        for (int n = 0; n < N; ++n) mx = fmaxf(mx, sc[n]);
        float sm = 0.f;
        for (int n = 0; n < N; ++n) { sc[n] = __expf(sc[n] - mx); sm += sc[n]; }
        float r = 1.f / sm;
        for (int n = 0; n < N; ++n) sc[n] *= r;
    }
    __syncthreads();
    // ws[j] = sum_n attn_n * cand[n,j]   (so out_d = vw[d,:] . ws)
    for (int j = tid; j < DM; j += 128) {
        float a = 0.f;
        for (int n = 0; n < N; ++n) a += sc[n] * cb[n * DM + j];
        ws[j] = a;
    }
    __syncthreads();
    { float a = 0.f; const float* w = vw + tid * DM;
      for (int j = 0; j < DM; ++j) a += ws[j] * w[j];
      ob[tid] = a; }
    __syncthreads();
    for (int j = tid; j < DM; j += 128) {
        float a = 0.f; const float* w = ow + j * DH;
        for (int d = 0; d < DH; ++d) a += ob[d] * w[d];
        ctx[b * DM + j] = cx[j] + a;
    }
}

// ---------------- scoring head input: concat(ctx[b], cand[b,n]) in f16 ----------------
__global__ void score_concat_kernel(const float* __restrict__ ctx, const float* __restrict__ cand,
                                    _Float16* __restrict__ o, int N, size_t total) {
    size_t i = (size_t)blockIdx.x * 256 + threadIdx.x;
    if (i >= total) return;
    int r = (int)(i / DM2), j = (int)(i % DM2);
    int b = r / N;
    float v = (j < DM) ? ctx[b * DM + j] : cand[(size_t)r * DM + (j - DM)];
    o[i] = (_Float16)v;
}

// ---------------- final score: out[r] = h[r,:] . w2 + b2 ----------------
__global__ void score2_kernel(const float* __restrict__ h, const float* __restrict__ w2,
                              const float* __restrict__ b2, float* __restrict__ out, int R) {
    int r = blockIdx.x * blockDim.x + threadIdx.x;
    if (r >= R) return;
    float s = 0.f;
    const float* a = h + (size_t)r * DM;
    for (int k = 0; k < DM; ++k) s += a[k] * w2[k];
    out[r] = s + b2[0];
}

extern "C" void kernel_launch(void* const* d_in, const int* in_sizes, int n_in,
                              void* d_out, int out_size, void* d_ws, size_t ws_size,
                              hipStream_t stream) {
    const float* events = (const float*)d_in[0];
    const int*   etypes = (const int*)d_in[1];
    const float* cand   = (const float*)d_in[2];
    const int*   plen   = (const int*)d_in[3];
    const float* temb   = (const float*)d_in[4];
    const float* ipw    = (const float*)d_in[5];
    const float* ipb    = (const float*)d_in[6];
    const float* fng    = (const float*)d_in[25];
    const float* fnb    = (const float*)d_in[26];
    const float* sw1    = (const float*)d_in[51];
    const float* sb1    = (const float*)d_in[52];
    const float* sw2    = (const float*)d_in[53];
    const float* sb2    = (const float*)d_in[54];

    const int Bn   = in_sizes[3];
    const int L    = in_sizes[0] / (Bn * DM);
    const int N    = in_sizes[2] / (Bn * DM);
    const int ROWS = Bn * L;

    // workspace carve-out
    char* ws = (char*)d_ws;
    size_t off = 0;
    auto alloc = [&](size_t bytes) -> void* {
        void* p = ws + off;
        off = (off + bytes + 255) & ~(size_t)255;
        return p;
    };
    float*    x    = (float*)alloc((size_t)ROWS * DM * 4);       // activations (residual stream)
    float*    xz   = (float*)alloc((size_t)ROWS * DM2 * 4);      // in_proj output
    float*    t384 = (float*)alloc((size_t)ROWS * DM * 4);       // x_ssm / y / xn / h
    _Float16* a16  = (_Float16*)alloc((size_t)ROWS * DM2 * 2);   // f16 GEMM activations
    _Float16* w16  = (_Float16*)alloc((size_t)DM * DM2 * 2);     // f16 weights (max 768x384)
    float*    xprm = (float*)alloc((size_t)ROWS * 9 * 4);        // x_proj output
    float*    ctx  = (float*)alloc((size_t)Bn * DM * 4);         // pooled context

    auto cvt = [&](const float* s, _Float16* d, size_t n) {
        f32_to_f16_kernel<<<dim3((unsigned)((n + 255) / 256)), 256, 0, stream>>>(s, d, n);
    };
    // M must be a multiple of 256 or have a tail handled by the wave-uniform guard (tile = 32 rows)
    auto gemm = [&](const _Float16* A, const _Float16* W, float* C, const float* bias,
                    const float* resid, int M, int Nn, int K, int act) {
        dim3 g((unsigned)(Nn / 32), (unsigned)((M + 255) / 256));
        wmma_gemm_kernel<<<g, 256, 0, stream>>>(A, W, C, bias, resid, M, Nn, K, act);
    };

    // ---- input projection ----
    embed_kernel<<<dim3((unsigned)(((size_t)ROWS * DM2 + 255) / 256)), 256, 0, stream>>>(
        events, etypes, temb, a16, ROWS);
    cvt(ipw, w16, (size_t)DM * DM2);
    gemm(a16, w16, x, ipb, nullptr, ROWS, DM, DM2, 0);

    // ---- SSM layers ----
    for (int li = 0; li < 2; ++li) {
        int pb = 7 + li * 9;
        const float* ng  = (const float*)d_in[pb + 0];
        const float* nb  = (const float*)d_in[pb + 1];
        const float* inw = (const float*)d_in[pb + 2];
        const float* cw  = (const float*)d_in[pb + 3];
        const float* cbv = (const float*)d_in[pb + 4];
        const float* xw  = (const float*)d_in[pb + 5];
        const float* la  = (const float*)d_in[pb + 6];
        const float* Dv  = (const float*)d_in[pb + 7];
        const float* ow  = (const float*)d_in[pb + 8];

        ln_kernel<<<ROWS, 128, 0, stream>>>(x, ng, nb, nullptr, a16);
        cvt(inw, w16, (size_t)DM2 * DM);
        gemm(a16, w16, xz, nullptr, nullptr, ROWS, DM2, DM, 0);
        conv_silu_kernel<<<dim3((unsigned)(((size_t)ROWS * DM + 255) / 256)), 256, 0, stream>>>(
            xz, cw, cbv, t384, ROWS, L);
        xparams_kernel<<<dim3((unsigned)(((size_t)ROWS * 9 + 255) / 256)), 256, 0, stream>>>(
            t384, xw, xprm, ROWS);
        scan_kernel<<<dim3((unsigned)((Bn * DM + 127) / 128)), 128, 0, stream>>>(
            xprm, la, Dv, t384, Bn, L);
        gate_kernel<<<dim3((unsigned)(((size_t)ROWS * DM + 255) / 256)), 256, 0, stream>>>(
            t384, xz, a16, (size_t)ROWS * DM);
        cvt(ow, w16, (size_t)DM * DM);
        gemm(a16, w16, x, nullptr, x, ROWS, DM, DM, 0);
    }

    // ---- final norm + pool ----
    ln_kernel<<<ROWS, 128, 0, stream>>>(x, fng, fnb, t384, nullptr);
    pool_kernel<<<dim3((unsigned)((Bn * DM + 127) / 128)), 128, 0, stream>>>(t384, plen, ctx, Bn, L);

    // ---- probes (sequential) ----
    for (int p = 0; p < 4; ++p) {
        int pb = 27 + p * 6;
        probe_kernel<<<Bn, 128, 0, stream>>>(
            ctx, cand,
            (const float*)d_in[pb + 0], (const float*)d_in[pb + 1],
            (const float*)d_in[pb + 2], (const float*)d_in[pb + 3],
            (const float*)d_in[pb + 4], (const float*)d_in[pb + 5], N);
    }

    // ---- scoring head ----
    const int R = Bn * N;
    score_concat_kernel<<<dim3((unsigned)(((size_t)R * DM2 + 255) / 256)), 256, 0, stream>>>(
        ctx, cand, a16, N, (size_t)R * DM2);
    cvt(sw1, w16, (size_t)DM * DM2);
    gemm(a16, w16, t384, sb1, nullptr, R, DM, DM2, 1);
    score2_kernel<<<dim3((unsigned)((R + 255) / 256)), 256, 0, stream>>>(
        t384, sw2, sb2, (float*)d_out, R);
}
